// FactoredInnerIteration_24507083391171
// MI455X (gfx1250) — matrix operand order repaired and checked by reading, not compiled, not run on hardware
//
#include <hip/hip_runtime.h>
#include <hip/hip_bf16.h>
#include <math.h>

// ---------------------------------------------------------------------------
// Problem constants (from reference): B=4, NV=2048, NC=8192, D=64, GD=8
// ---------------------------------------------------------------------------
#define NVV  2048
#define NCC  8192
#define NBAT 4
#define DD   64
#define GDIM 8
#define RV   (NBAT * NVV)   // 8192  flat variable rows
#define RC   (NBAT * NCC)   // 32768 flat clause rows

typedef __bf16 bf16_t;
typedef __bf16 v16bf __attribute__((ext_vector_type(16)));
typedef float  v8f   __attribute__((ext_vector_type(8)));

union F16x16 { v16bf v; uint4 u[2]; };
union F8x16  { uint4 u; bf16_t h[8]; };

__device__ __forceinline__ v8f wmma_bf16(v16bf a, v16bf b, v8f c) {
  // D(16x16,f32) = A(16x32,bf16) * B(32x16,bf16) + C
  return __builtin_amdgcn_wmma_f32_16x16x32_bf16(
      /*neg_a=*/false, a, /*neg_b=*/false, b,
      /*c_mod=*/(short)0, c, /*reuse_a=*/false, /*reuse_b=*/false);
}

__device__ __forceinline__ v8f zero8() {
  v8f z;
#pragma unroll
  for (int r = 0; r < 8; ++r) z[r] = 0.f;
  return z;
}

__device__ __forceinline__ float sigmoidf_(float x) {
  return 1.f / (1.f + __expf(-x));
}

// ---- A fragment (16x32 bf16, M x K) from f32 row-major tile -----------------
// lane L<16 : row M=L, K chunks [0..7] (v0..3) and [16..23] (v4..7)
// lane L>=16: row M=L-16, K chunks [8..15] and [24..31]
__device__ __forceinline__ v16bf a_frag_f32(const float* base, int ld,
                                            int kbase, int lane) {
  const int m  = lane & 15;
  const int k0 = (lane & 16) ? 8 : 0;
  const float* p = base + (long)m * ld + kbase + k0;
  v16bf a;
#pragma unroll
  for (int t = 0; t < 8; ++t) {
    a[t]     = (bf16_t)p[t];
    a[t + 8] = (bf16_t)p[t + 16];
  }
  return a;
}

// ---- A fragments (pos/neg clip) from f32 adjacency tile ---------------------
__device__ __forceinline__ void a_frag_adj(const float* base, long ld, int lane,
                                           v16bf& ap, v16bf& an) {
  const int m  = lane & 15;
  const int k0 = (lane & 16) ? 8 : 0;
  const float* p = base + (long)m * ld + k0;
#pragma unroll
  for (int t = 0; t < 8; ++t) {
    float x0 = p[t], x1 = p[t + 16];
    ap[t]     = (bf16_t)fmaxf(x0, 0.f);
    an[t]     = (bf16_t)fmaxf(-x0, 0.f);
    ap[t + 8] = (bf16_t)fmaxf(x1, 0.f);
    an[t + 8] = (bf16_t)fmaxf(-x1, 0.f);
  }
}

// ---- A fragment from bf16 row-major LDS tile (16 x 64) ----------------------
__device__ __forceinline__ v16bf a_frag_lds(const bf16_t* tile, int kbase,
                                            int lane) {
  const int m  = lane & 15;
  const int k0 = (lane & 16) ? 8 : 0;
  const bf16_t* p = tile + m * DD + kbase + k0;
  F16x16 f;
  f.u[0] = *(const uint4*)p;         // K = k0 .. k0+7
  f.u[1] = *(const uint4*)(p + 16);  // K = k0+16 .. k0+23
  return f.v;
}

// ---- A fragment for ground vars: 16 x 8 real, zero-padded to 16 x 32 --------
__device__ __forceinline__ v16bf a_frag_g(const float* gt, int lane) {
  const int  m  = lane & 15;
  const bool lo = (lane & 16) == 0;   // lanes 0-15 carry K=0..7 (real part)
  v16bf a;
#pragma unroll
  for (int t = 0; t < 8; ++t) {
    a[t]     = (bf16_t)(lo ? gt[m * GDIM + t] : 0.f);
    a[t + 8] = (bf16_t)0.f;
  }
  return a;
}

// ---- B fragment (32x16 bf16, K x N): lanes 0-15 K=0..15, lanes 16-31 K=16..31
// from K-contiguous bf16 memory (element (k,n) at base[n*ldn + k])
__device__ __forceinline__ v16bf b_frag_kmajor(const bf16_t* base, long ldn,
                                               int nbase, int lane) {
  const int n  = nbase + (lane & 15);
  const int kb = (lane & 16) ? 16 : 0;
  const bf16_t* p = base + (long)n * ldn + kb;
  F16x16 f;
  f.u[0] = *(const uint4*)p;
  f.u[1] = *(const uint4*)(p + 8);
  return f.v;
}

// ---- B fragment from LDS slab laid out as [n][32 k] -------------------------
__device__ __forceinline__ v16bf b_frag_lds(const bf16_t* lds, int nb, int lane) {
  const int n  = nb * 16 + (lane & 15);
  const int kb = (lane & 16) ? 8 : 0;
  const bf16_t* p = lds + n * 32 + kb;
  F16x16 f;
  f.u[0] = *(const uint4*)p;
  f.u[1] = *(const uint4*)(p + 16);
  return f.v;
}

// ---- B fragment from f32 weight W[n, k] (row-major, (X@W^T) convention) -----
__device__ __forceinline__ v16bf b_frag_w(const float* W, int ldk, int kofs,
                                          int kbase, int nbase, int lane) {
  const int n  = nbase + (lane & 15);
  const int kb = kbase + ((lane & 16) ? 8 : 0);
  const float* p = W + (long)n * ldk + kofs + kb;
  v16bf b;
#pragma unroll
  for (int t = 0; t < 16; ++t) b[t] = (bf16_t)p[t < 8 ? t : t + 8];
  return b;
}

// ---- B fragment for Wg's ground part: real K=0..7, zero-padded to 32 --------
__device__ __forceinline__ v16bf b_frag_wg_g(const float* Wg, int nbase,
                                             int lane) {
  const int  n  = nbase + (lane & 15);
  const bool lo = (lane & 16) == 0;
  v16bf b;
#pragma unroll
  for (int t = 0; t < 16; ++t) {
    float x = (lo && t < GDIM) ? Wg[n * (GDIM + DD) + t] : 0.f;
    b[t] = (bf16_t)x;
  }
  return b;
}

// ===========================================================================
// Kernel 1/3: dual projection  out{0,1}T[64, M] = bf16( (X @ W{0,1}^T)^T )
// Block = 256 threads = 8 waves, each wave owns 16 rows x all 64 cols.
// ===========================================================================
__global__ __launch_bounds__(256) void proj2_kernel(
    const float* __restrict__ X,   // [M, 64] row-major f32
    const float* __restrict__ W0,  // [64, 64]
    const float* __restrict__ W1,  // [64, 64]
    bf16_t* __restrict__ o0T,      // [64, M] K-contiguous bf16
    bf16_t* __restrict__ o1T,
    int M) {
  const int lane = threadIdx.x & 31;
  const int wave = threadIdx.x >> 5;
  const long m0  = (long)blockIdx.x * 128 + wave * 16;

  const float* Xt = X + m0 * DD;
  v16bf a0 = a_frag_f32(Xt, DD, 0, lane);
  v16bf a1 = a_frag_f32(Xt, DD, 32, lane);

  const int n    = lane & 15;
  const int moff = (lane & 16) ? 8 : 0;

#pragma unroll
  for (int w = 0; w < 2; ++w) {
    const float* W = w ? W1 : W0;
    bf16_t* oT = w ? o1T : o0T;
#pragma unroll
    for (int nb = 0; nb < 4; ++nb) {
      v8f acc = zero8();
      acc = wmma_bf16(a0, b_frag_w(W, DD, 0, 0, nb * 16, lane), acc);
      acc = wmma_bf16(a1, b_frag_w(W, DD, 0, 32, nb * 16, lane), acc);
      F8x16 s;
#pragma unroll
      for (int r = 0; r < 8; ++r) s.h[r] = (bf16_t)acc[r];
      // transposed store: column n, 8 consecutive rows -> one 16B store
      *(uint4*)&oT[(long)(nb * 16 + n) * M + m0 + moff] = s.u;
    }
  }
}

// ===========================================================================
// Kernel 2/4: adjacency matmul
//   out[b,m,:] = relu( clip+(adj[b,m,:]) @ pos[b] + clip-(adj[b,m,:]) @ neg[b]
//                      + bias )
// adj streamed f32 (clip->bf16 in regs), pos/neg panels staged in LDS per
// K-slab of 32 and shared by all 8 waves. 8 WMMAs per wave per K-step.
// ===========================================================================
__global__ __launch_bounds__(256) void adj_matmul_kernel(
    const float* __restrict__ adj,  // [NBAT, M, K] row-major f32
    const bf16_t* __restrict__ pT,  // [64, NBAT*K] K-contiguous bf16
    const bf16_t* __restrict__ nT,  // [64, NBAT*K]
    const float* __restrict__ bias, // [64]
    float* __restrict__ out,        // [NBAT, M, 64] f32
    int M, int K, long ldB) {
  __shared__ __align__(16) bf16_t lds_p[DD * 32];
  __shared__ __align__(16) bf16_t lds_n[DD * 32];

  const int b    = blockIdx.y;
  const int lane = threadIdx.x & 31;
  const int wave = threadIdx.x >> 5;
  const long m0  = (long)blockIdx.x * 128 + wave * 16;

  const float*  A  = adj + ((long)b * M + m0) * K;
  const bf16_t* pB = pT + (long)b * K;
  const bf16_t* nB = nT + (long)b * K;

  v8f acc[4];
#pragma unroll
  for (int nb = 0; nb < 4; ++nb) acc[nb] = zero8();

  const int tn = threadIdx.x >> 2;        // 0..63 : column n
  const int tc = (threadIdx.x & 3) * 8;   // 0,8,16,24 : k chunk

  for (int k0 = 0; k0 < K; k0 += 32) {
    __syncthreads();  // previous slab fully consumed
    *(uint4*)&lds_p[tn * 32 + tc] = *(const uint4*)&pB[(long)tn * ldB + k0 + tc];
    *(uint4*)&lds_n[tn * 32 + tc] = *(const uint4*)&nB[(long)tn * ldB + k0 + tc];
    __syncthreads();

    v16bf ap, an;
    a_frag_adj(A + k0, K, lane, ap, an);
    if (k0 + 32 < K) {
      const float* pf =
          A + (long)(lane & 15) * K + k0 + 32 + ((lane & 16) ? 8 : 0);
      __builtin_prefetch(pf, 0, 0);  // global_prefetch_b8 for next A slab
    }
#pragma unroll
    for (int nb = 0; nb < 4; ++nb) {
      acc[nb] = wmma_bf16(ap, b_frag_lds(lds_p, nb, lane), acc[nb]);
      acc[nb] = wmma_bf16(an, b_frag_lds(lds_n, nb, lane), acc[nb]);
    }
  }

  const int n    = lane & 15;
  const int moff = (lane & 16) ? 8 : 0;
#pragma unroll
  for (int nb = 0; nb < 4; ++nb) {
    const float bv = bias[nb * 16 + n];
#pragma unroll
    for (int r = 0; r < 8; ++r) {
      float v = fmaxf(acc[nb][r] + bv, 0.f);
      out[((long)b * M + m0 + moff + r) * DD + nb * 16 + n] = v;
    }
  }
}

// ===========================================================================
// Kernel 5: fused ground combiner + GRU. Each wave owns 16 rows.
//   av = tanh([g, v_emb] @ Wg^T + bg)
//   z  = sigmoid(av@Wz^T + prev@Uz^T + bz)
//   r  = sigmoid(av@Wr^T + prev@Ur^T + br)
//   h~ = tanh(av@Wh^T + (r*prev)@Uh^T + bh)
//   h  = (1-z)*prev + z*h~
// av and r*prev transpose (D-layout -> A-layout) bounces through LDS.
// ===========================================================================
__global__ __launch_bounds__(256) void gru_kernel(
    const float* __restrict__ g,   // [RV, GDIM]
    const float* __restrict__ ve,  // [RV, 64]  v_emb
    const float* __restrict__ pv,  // [RV, 64]  prev (= variables)
    const float* __restrict__ Wg,  // [64, 72]
    const float* __restrict__ Wz, const float* __restrict__ Uz,
    const float* __restrict__ Wr, const float* __restrict__ Ur,
    const float* __restrict__ Wh, const float* __restrict__ Uh,
    const float* __restrict__ bg, const float* __restrict__ bz,
    const float* __restrict__ br, const float* __restrict__ bh,
    float* __restrict__ out) {   // [RV, 64]
  __shared__ __align__(16) bf16_t s_av[8 * 16 * DD];
  __shared__ __align__(16) bf16_t s_rp[8 * 16 * DD];

  const int lane = threadIdx.x & 31;
  const int wave = threadIdx.x >> 5;
  const long m0  = (long)blockIdx.x * 128 + wave * 16;
  bf16_t* av_t = s_av + wave * 16 * DD;
  bf16_t* rp_t = s_rp + wave * 16 * DD;

  const int n    = lane & 15;
  const int moff = (lane & 16) ? 8 : 0;

  v16bf a_g  = a_frag_g(g + m0 * GDIM, lane);
  v16bf a_e0 = a_frag_f32(ve + m0 * DD, DD, 0, lane);
  v16bf a_e1 = a_frag_f32(ve + m0 * DD, DD, 32, lane);
  v16bf a_p0 = a_frag_f32(pv + m0 * DD, DD, 0, lane);
  v16bf a_p1 = a_frag_f32(pv + m0 * DD, DD, 32, lane);

  // prev in C/D layout for the pointwise blend
  float pvd[4][8];
#pragma unroll
  for (int nb = 0; nb < 4; ++nb)
#pragma unroll
    for (int r = 0; r < 8; ++r)
      pvd[nb][r] = pv[(m0 + moff + r) * DD + nb * 16 + n];

  // ---- av = tanh([g, v_emb] @ Wg^T + bg), K = 8 (padded) + 64 -------------
#pragma unroll
  for (int nb = 0; nb < 4; ++nb) {
    v8f acc = zero8();
    acc = wmma_bf16(a_g,  b_frag_wg_g(Wg, nb * 16, lane), acc);
    acc = wmma_bf16(a_e0, b_frag_w(Wg, GDIM + DD, GDIM, 0,  nb * 16, lane), acc);
    acc = wmma_bf16(a_e1, b_frag_w(Wg, GDIM + DD, GDIM, 32, nb * 16, lane), acc);
    const float bb = bg[nb * 16 + n];
#pragma unroll
    for (int r = 0; r < 8; ++r)
      av_t[(moff + r) * DD + nb * 16 + n] = (bf16_t)tanhf(acc[r] + bb);
  }
  __syncthreads();
  v16bf a_av0 = a_frag_lds(av_t, 0, lane);
  v16bf a_av1 = a_frag_lds(av_t, 32, lane);

  // ---- r gate; rp = r * prev -> LDS ---------------------------------------
#pragma unroll
  for (int nb = 0; nb < 4; ++nb) {
    v8f acc = zero8();
    acc = wmma_bf16(a_av0, b_frag_w(Wr, DD, 0, 0,  nb * 16, lane), acc);
    acc = wmma_bf16(a_av1, b_frag_w(Wr, DD, 0, 32, nb * 16, lane), acc);
    acc = wmma_bf16(a_p0,  b_frag_w(Ur, DD, 0, 0,  nb * 16, lane), acc);
    acc = wmma_bf16(a_p1,  b_frag_w(Ur, DD, 0, 32, nb * 16, lane), acc);
    const float bb = br[nb * 16 + n];
#pragma unroll
    for (int r = 0; r < 8; ++r) {
      float rg = sigmoidf_(acc[r] + bb);
      rp_t[(moff + r) * DD + nb * 16 + n] = (bf16_t)(rg * pvd[nb][r]);
    }
  }

  // ---- z gate (kept in registers) -----------------------------------------
  float zg[4][8];
#pragma unroll
  for (int nb = 0; nb < 4; ++nb) {
    v8f acc = zero8();
    acc = wmma_bf16(a_av0, b_frag_w(Wz, DD, 0, 0,  nb * 16, lane), acc);
    acc = wmma_bf16(a_av1, b_frag_w(Wz, DD, 0, 32, nb * 16, lane), acc);
    acc = wmma_bf16(a_p0,  b_frag_w(Uz, DD, 0, 0,  nb * 16, lane), acc);
    acc = wmma_bf16(a_p1,  b_frag_w(Uz, DD, 0, 32, nb * 16, lane), acc);
    const float bb = bz[nb * 16 + n];
#pragma unroll
    for (int r = 0; r < 8; ++r) zg[nb][r] = sigmoidf_(acc[r] + bb);
  }

  __syncthreads();
  v16bf a_r0 = a_frag_lds(rp_t, 0, lane);
  v16bf a_r1 = a_frag_lds(rp_t, 32, lane);

  // ---- h_tilda and blended output -----------------------------------------
#pragma unroll
  for (int nb = 0; nb < 4; ++nb) {
    v8f acc = zero8();
    acc = wmma_bf16(a_av0, b_frag_w(Wh, DD, 0, 0,  nb * 16, lane), acc);
    acc = wmma_bf16(a_av1, b_frag_w(Wh, DD, 0, 32, nb * 16, lane), acc);
    acc = wmma_bf16(a_r0,  b_frag_w(Uh, DD, 0, 0,  nb * 16, lane), acc);
    acc = wmma_bf16(a_r1,  b_frag_w(Uh, DD, 0, 32, nb * 16, lane), acc);
    const float bb = bh[nb * 16 + n];
#pragma unroll
    for (int r = 0; r < 8; ++r) {
      float ht = tanhf(acc[r] + bb);
      float z  = zg[nb][r];
      out[(m0 + moff + r) * DD + nb * 16 + n] =
          (1.f - z) * pvd[nb][r] + z * ht;
    }
  }
}

// ===========================================================================
extern "C" void kernel_launch(void* const* d_in, const int* in_sizes, int n_in,
                              void* d_out, int out_size, void* d_ws,
                              size_t ws_size, hipStream_t stream) {
  (void)in_sizes; (void)n_in; (void)out_size; (void)ws_size;

  const float* variables = (const float*)d_in[0];   // [4,2048,64]
  const float* ground    = (const float*)d_in[1];   // [4,2048,8]
  const float* c_mat     = (const float*)d_in[2];   // [4,8192,2048]
  const float* v_mat     = (const float*)d_in[3];   // [4,2048,8192]
  const float* c_block   = (const float*)d_in[4];   // [2,64,64]
  const float* v_block   = (const float*)d_in[5];   // [2,64,64]
  const float* vb        = (const float*)d_in[6];   // [64]
  const float* cb        = (const float*)d_in[7];   // [64]
  const float* Wg        = (const float*)d_in[8];   // [64,72]
  const float* bg        = (const float*)d_in[9];   // [64]
  const float* Wz        = (const float*)d_in[10];
  const float* Uz        = (const float*)d_in[11];
  const float* bz        = (const float*)d_in[12];
  const float* Wr        = (const float*)d_in[13];
  const float* Ur        = (const float*)d_in[14];
  const float* br        = (const float*)d_in[15];
  const float* Wh        = (const float*)d_in[16];
  const float* Uh        = (const float*)d_in[17];
  const float* bh        = (const float*)d_in[18];
  float* out = (float*)d_out;                       // [4, 2048*64, 1] flat

  // workspace carve-up (~20 MB, 16B aligned throughout)
  char* ws = (char*)d_ws;
  bf16_t* posVT = (bf16_t*)ws; ws += (size_t)DD * RV * sizeof(bf16_t);
  bf16_t* negVT = (bf16_t*)ws; ws += (size_t)DD * RV * sizeof(bf16_t);
  float*  cbuf  = (float*)ws;  ws += (size_t)RC * DD * sizeof(float);
  bf16_t* posCT = (bf16_t*)ws; ws += (size_t)DD * RC * sizeof(bf16_t);
  bf16_t* negCT = (bf16_t*)ws; ws += (size_t)DD * RC * sizeof(bf16_t);
  float*  vemb  = (float*)ws;  ws += (size_t)RV * DD * sizeof(float);

  // 1) literal projections of variables (pos/neg), transposed bf16 panels
  proj2_kernel<<<RV / 128, 256, 0, stream>>>(
      variables, c_block, c_block + DD * DD, posVT, negVT, RV);

  // 2) clause-side adjacency matmul: c = relu(pos@posV + neg@negV + cb)
  adj_matmul_kernel<<<dim3(NCC / 128, NBAT), 256, 0, stream>>>(
      c_mat, posVT, negVT, cb, cbuf, NCC, NVV, (long)RV);

  // 3) literal projections of clauses
  proj2_kernel<<<RC / 128, 256, 0, stream>>>(
      cbuf, v_block, v_block + DD * DD, posCT, negCT, RC);

  // 4) variable-side adjacency matmul: v_emb = relu(... + vb)
  adj_matmul_kernel<<<dim3(NVV / 128, NBAT), 256, 0, stream>>>(
      v_mat, posCT, negCT, vb, vemb, NVV, NCC, (long)RC);

  // 5) fused ground combiner + GRU -> output
  gru_kernel<<<RV / 128, 256, 0, stream>>>(
      ground, vemb, variables, Wg, Wz, Uz, Wr, Ur, Wh, Uh,
      bg, bz, br, bh, out);
}
